// GraphEncoder_20727512170754
// MI455X (gfx1250) — matrix operand order repaired and checked by reading, not compiled, not run on hardware
//
#include <hip/hip_runtime.h>

// ---------------- types ----------------
typedef __attribute__((ext_vector_type(16))) __bf16 v16bf;
typedef __attribute__((ext_vector_type(8)))  float  v8f;

union FragU { v16bf v; unsigned int u[8]; };

__device__ __forceinline__ unsigned short f2bf(float f) {
    unsigned int u = __float_as_uint(f);
    unsigned int r = u + 0x7FFFu + ((u >> 16) & 1u);   // round-to-nearest-even
    return (unsigned short)(r >> 16);
}

// ---------------- constants ----------------
#define BSZ   32
#define MEc   384
#define Rc    128
#define Nc    512
#define Hc    500
#define Hp    512
#define F4    2000
#define F4p   2048
#define NH    4
#define DHp   128
#define MTOT  (BSZ * Nc)          // 16384

#define MODE_F32        0
#define MODE_BF16       1
#define MODE_BF16_VT    2
#define MODE_BF16_PRELU 3

// =======================================================================
// Weight prep: transpose to Bt[c][k] (bf16, padded with zeros), optional
// per-head dh-pad remap (c -> head*125 + (c%128), invalid if c%128 >= 125)
// =======================================================================
__global__ void prep_weight_kernel(const float* __restrict__ W, int Ksrc, int Csrc,
                                   unsigned short* __restrict__ out, int Cp, int Kp,
                                   int headpad) {
    long long idx = (long long)blockIdx.x * 256 + threadIdx.x;
    long long tot = (long long)Cp * Kp;
    if (idx >= tot) return;
    int c = (int)(idx / Kp);
    int k = (int)(idx % Kp);
    int src_c = c;
    bool valid = true;
    if (headpad) { int d = c & 127; src_c = (c >> 7) * 125 + d; valid = (d < 125); }
    float v = 0.f;
    if (valid && src_c < Csrc && k < Ksrc) v = W[(long long)k * Csrc + src_c];
    out[idx] = f2bf(v);
}

// =======================================================================
// Node assembly: entities then relation embeddings, zero past size.
// Writes fp32 (ld=500) and bf16 (ld=512, zero pads).
// =======================================================================
__global__ void build_nodes_kernel(const float* __restrict__ ent_vec,
                                   const int* __restrict__ entity_num,
                                   const int* __restrict__ rels,
                                   const float* __restrict__ rel_embed,
                                   float* __restrict__ nodes_f,
                                   unsigned short* __restrict__ nodes_h) {
    long long idx = (long long)blockIdx.x * 256 + threadIdx.x; // B*N*512
    int h = (int)(idx & 511);
    long long rn = idx >> 9;          // b*512 + n
    int n = (int)(rn & 511);
    int b = (int)(rn >> 9);
    int en = entity_num[b];
    float v = 0.f;
    if (h < Hc) {
        if (n < en)            v = ent_vec[((long long)b * MEc + n) * Hc + h];
        else if (n < en + Rc)  { int rid = rels[b * Rc + (n - en)];
                                 v = rel_embed[(long long)rid * Hc + h]; }
        nodes_f[rn * Hc + h] = v;
    }
    nodes_h[rn * Hp + h] = f2bf(v);
}

// =======================================================================
// Batched WMMA GEMM: C[M x Npad] = A[M x K](bf16,row) * Bt[Npad x K](bf16,row)^T
// Per-z offsets: off = (z/bdiv)*so + (z%bdiv)*si  (elements).
// Block 256 = 8 waves; wave tile 32(M) x 64(N): 2 A frags x 4 B frags
// -> 8 v_wmma per K-step (K stepped by 32).
// =======================================================================
__global__ __launch_bounds__(256) void gemm_wmma_kernel(
    const unsigned short* __restrict__ A,  long long aso, long long asi, int lda,
    const unsigned short* __restrict__ Bt, long long bso, long long bsi, int ldb,
    const float* __restrict__ bias, const float* __restrict__ prelu,
    float* __restrict__ outf, unsigned short* __restrict__ outh,
    long long oso, long long osi, int ldo,
    int M, int Ncols, int K, int bdiv, int mode) {

    const int lane = threadIdx.x & 31;
    const int wave = threadIdx.x >> 5;
    const int m0 = blockIdx.x * 256 + wave * 32;
    const int n0 = blockIdx.y * 64;
    const int z  = blockIdx.z;
    if (m0 >= M) return;

    const long long aoff = (long long)(z / bdiv) * aso + (long long)(z % bdiv) * asi;
    const long long boff = (long long)(z / bdiv) * bso + (long long)(z % bdiv) * bsi;
    const long long ooff = (long long)(z / bdiv) * oso + (long long)(z % bdiv) * osi;

    const int mr = lane & 15;        // A: M row, B: N row, C: col
    const int hs = lane >> 4;        // half-wave select

    v8f zero = {0.f,0.f,0.f,0.f,0.f,0.f,0.f,0.f};
    v8f acc[2][4];
#pragma unroll
    for (int i = 0; i < 2; ++i)
#pragma unroll
        for (int j = 0; j < 4; ++j) acc[i][j] = zero;

    const unsigned short* ap0 = A  + aoff + (long long)(m0 + mr) * lda;
    const unsigned short* ap1 = ap0 + (long long)16 * lda;
    const unsigned short* bp0 = Bt + boff + (long long)(n0 + mr) * ldb;

    for (int k = 0; k < K; k += 32) {
        FragU fa0, fa1;
#pragma unroll
        for (int v = 0; v < 8; ++v) {
            int kk = k + ((v & 4) << 2) + (hs << 3) + ((v & 3) << 1);
            fa0.u[v] = *(const unsigned int*)(ap0 + kk);
            fa1.u[v] = *(const unsigned int*)(ap1 + kk);
        }
#pragma unroll
        for (int j = 0; j < 4; ++j) {
            FragU fb;
            const unsigned short* bp = bp0 + (long long)(j * 16) * ldb + k;
#pragma unroll
            for (int v = 0; v < 8; ++v) fb.u[v] = *(const unsigned int*)(bp + ((v & 4) << 2) + (hs << 3) + ((v & 3) << 1));
            acc[0][j] = __builtin_amdgcn_wmma_f32_16x16x32_bf16(
                false, fa0.v, false, fb.v, (short)0, acc[0][j], false, false);
            acc[1][j] = __builtin_amdgcn_wmma_f32_16x16x32_bf16(
                false, fa1.v, false, fb.v, (short)0, acc[1][j], false, false);
        }
    }

#pragma unroll
    for (int i = 0; i < 2; ++i) {
#pragma unroll
        for (int j = 0; j < 4; ++j) {
#pragma unroll
            for (int v = 0; v < 8; ++v) {
                int row = m0 + i * 16 + (hs << 3) + v;
                int col = n0 + j * 16 + mr;
                float val = acc[i][j][v];
                if (mode == MODE_F32) {
                    if (col < Ncols)
                        outf[ooff + (long long)row * ldo + col] = val + (bias ? bias[col] : 0.f);
                } else if (mode == MODE_BF16) {
                    if (col < Ncols)
                        outh[ooff + (long long)row * ldo + col] = f2bf(val + (bias ? bias[col] : 0.f));
                } else if (mode == MODE_BF16_VT) {
                    // vT[b][head][d][i]: ((b*512 + col)*512 + i), b=row>>9, i=row&511
                    long long o = ((long long)((row >> 9) * 512 + col)) * 512 + (row & 511);
                    outh[o] = f2bf(val);
                } else { // MODE_BF16_PRELU
                    float hh = 0.f;
                    if (col < Ncols) {
                        hh = val + (bias ? bias[col] : 0.f);
                        if (hh < 0.f) hh *= prelu[col];
                    }
                    outh[ooff + (long long)row * ldo + col] = f2bf(hh);
                }
            }
        }
    }
}

// =======================================================================
// Mask + scale + softmax over rows of s[z][i][:], write bf16 probs
// =======================================================================
__global__ __launch_bounds__(256) void softmax_mask_kernel(
    const float* __restrict__ s, const int* __restrict__ adj,
    unsigned short* __restrict__ p) {
    const float scale = 0.08944271909999159f; // 1/sqrt(125)
    __shared__ float red[256];
    int bid = blockIdx.x;            // z*512 + i
    int z = bid >> 9, i = bid & 511;
    int b = z >> 2;
    int t = threadIdx.x;
    const float* srow = s + (long long)bid * 512;
    const int*   arow = adj + ((long long)b * 512 + i) * 512;
    int j0 = t, j1 = t + 256;
    float x0 = arow[j0] ? srow[j0] * scale : -1e9f;
    float x1 = arow[j1] ? srow[j1] * scale : -1e9f;
    red[t] = fmaxf(x0, x1); __syncthreads();
    for (int st = 128; st > 0; st >>= 1) { if (t < st) red[t] = fmaxf(red[t], red[t + st]); __syncthreads(); }
    float mx = red[0]; __syncthreads();
    float e0 = __expf(x0 - mx), e1 = __expf(x1 - mx);
    red[t] = e0 + e1; __syncthreads();
    for (int st = 128; st > 0; st >>= 1) { if (t < st) red[t] += red[t + st]; __syncthreads(); }
    float inv = 1.0f / red[0];
    unsigned short* prow = p + (long long)bid * 512;
    prow[j0] = f2bf(e0 * inv);
    prow[j1] = f2bf(e1 * inv);
}

// =======================================================================
// LayerNorm over 500 channels: x = x1 (+x2), write fp32 (ld=500) and
// bf16 padded (ld=512, zero pads). Safe when outf aliases x2.
// =======================================================================
__global__ __launch_bounds__(512) void layernorm_kernel(
    const float* __restrict__ x1, const float* __restrict__ x2,
    const float* __restrict__ g, const float* __restrict__ bb,
    float* __restrict__ outf, unsigned short* __restrict__ outh) {
    __shared__ float red[512];
    const int row = blockIdx.x;
    const int h = threadIdx.x;
    float v = 0.f;
    if (h < Hc) {
        v = x1[(long long)row * Hc + h];
        if (x2) v += x2[(long long)row * Hc + h];
    }
    red[h] = v; __syncthreads();
    for (int s = 256; s > 0; s >>= 1) { if (h < s) red[h] += red[h + s]; __syncthreads(); }
    const float mu = red[0] * (1.f / 500.f); __syncthreads();
    const float d = (h < Hc) ? (v - mu) : 0.f;
    red[h] = d * d; __syncthreads();
    for (int s = 256; s > 0; s >>= 1) { if (h < s) red[h] += red[h + s]; __syncthreads(); }
    const float inv = rsqrtf(red[0] * (1.f / 500.f) + 1e-5f);
    float y = 0.f;
    if (h < Hc) {
        y = d * inv * g[h] + bb[h];
        outf[(long long)row * Hc + h] = y;
    }
    outh[(long long)row * Hp + h] = (h < Hc) ? f2bf(y) : (unsigned short)0;
}

// =======================================================================
// Outputs: glob (B x 500) | node_emb (B x 512 x 500, masked) | mask (B x 512)
// =======================================================================
__global__ void node_emb_kernel(const float* __restrict__ nodes,
                                const int* __restrict__ entity_num,
                                float* __restrict__ out) {
    long long idx = (long long)blockIdx.x * 256 + threadIdx.x;
    if (idx >= (long long)BSZ * Nc * Hc) return;
    int h = (int)(idx % Hc);
    long long rn = idx / Hc;
    int n = (int)(rn % Nc);
    int b = (int)(rn / Nc);
    int sz = entity_num[b] + Rc;
    float v = (n < sz) ? nodes[rn * Hc + h] : 0.f;
    out[16000 + idx] = v;
}

__global__ void glob_mask_kernel(const float* __restrict__ nodes,
                                 const int* __restrict__ entity_num,
                                 float* __restrict__ out) {
    int t = blockIdx.x * 256 + threadIdx.x;
    if (t < BSZ * Nc) {
        int b = t >> 9, n = t & 511;
        int sz = entity_num[b] + Rc;
        out[16000 + BSZ * Nc * Hc + t] = (n <= sz) ? 1.0f : 0.0f;
    }
    if (t < BSZ * Hc) {
        int b = t / Hc, h = t % Hc;
        int en = entity_num[b];
        out[t] = nodes[((long long)b * Nc + en) * Hc + h];
    }
}

// =======================================================================
// Host side
// =======================================================================
static inline void gemm(hipStream_t st,
                        const void* A,  long long aso, long long asi, int lda,
                        const void* Bt, long long bso, long long bsi, int ldb,
                        const float* bias, const float* prelu,
                        void* out, long long oso, long long osi, int ldo,
                        int M, int Ncol, int Npad, int K, int bdiv, int nb, int mode) {
    dim3 g(M / 256, Npad / 64, nb);
    gemm_wmma_kernel<<<g, 256, 0, st>>>(
        (const unsigned short*)A, aso, asi, lda,
        (const unsigned short*)Bt, bso, bsi, ldb,
        bias, prelu,
        (mode == MODE_F32) ? (float*)out : nullptr,
        (mode == MODE_F32) ? nullptr : (unsigned short*)out,
        oso, osi, ldo, M, Ncol, K, bdiv, mode);
}

extern "C" void kernel_launch(void* const* d_in, const int* in_sizes, int n_in,
                              void* d_out, int out_size, void* d_ws, size_t ws_size,
                              hipStream_t stream) {
    (void)in_sizes; (void)n_in; (void)out_size; (void)ws_size;

    const float* ent_vec    = (const float*)d_in[0];
    const int*   entity_num = (const int*)d_in[1];
    const int*   rels       = (const int*)d_in[2];
    const int*   adj        = (const int*)d_in[3];
    const float* rel_embed  = (const float*)d_in[4];
    const float* Wq = (const float*)d_in[5];
    const float* Wk = (const float*)d_in[7];
    const float* Wv = (const float*)d_in[9];
    const float* Wo = (const float*)d_in[11];
    const float* bo = (const float*)d_in[12];
    const float* W1 = (const float*)d_in[13];
    const float* b1 = (const float*)d_in[14];
    const float* W2 = (const float*)d_in[15];
    const float* b2 = (const float*)d_in[16];
    const float* ln1_g = (const float*)d_in[17];
    const float* ln1_b = (const float*)d_in[18];
    const float* ln2_g = (const float*)d_in[19];
    const float* ln2_b = (const float*)d_in[20];
    const float* prelu_a = (const float*)d_in[21];
    float* out = (float*)d_out;

    char* ws = (char*)d_ws;
    const size_t OFF_A = 0;                       // nodes / t  fp32 (32.77 MB)
    const size_t OFF_B = 32768000;                // nodes / t  bf16 padded (16.78 MB)
    const size_t OFF_C = OFF_B + 16777216;        // q / ctx bf16
    const size_t OFF_D = OFF_C + 16777216;        // k bf16
    const size_t OFF_E = OFF_D + 16777216;        // vT bf16
    const size_t OFF_F = OFF_E + 16777216;        // scores fp32 / o / f (134.2 MB)
    const size_t OFF_G = OFF_F + 134217728;       // probs / h1 bf16 (67.1 MB)
    const size_t OFF_W = OFF_G + 67108864;        // weights (12.6 MB)

    float*          Af = (float*)(ws + OFF_A);
    unsigned short* Bh = (unsigned short*)(ws + OFF_B);
    unsigned short* Ch = (unsigned short*)(ws + OFF_C);
    unsigned short* Dh = (unsigned short*)(ws + OFF_D);
    unsigned short* Eh = (unsigned short*)(ws + OFF_E);
    float*          Ff = (float*)(ws + OFF_F);
    unsigned short* Gh = (unsigned short*)(ws + OFF_G);

    // ---- weight prep (bf16, transposed, zero-padded) ----
    for (int l = 0; l < 2; ++l) {
        unsigned short* wqT = (unsigned short*)(ws + OFF_W + (size_t)l * 6291456);
        unsigned short* wkT = wqT + 262144;
        unsigned short* wvT = wkT + 262144;
        unsigned short* woT = wvT + 262144;
        unsigned short* w1T = woT + 262144;     // 2048 x 512
        unsigned short* w2T = w1T + 1048576;    // 512 x 2048
        prep_weight_kernel<<<1024, 256, 0, stream>>>(Wq + (size_t)l*Hc*Hc, Hc, Hc,  wqT, Hp, Hp, 1);
        prep_weight_kernel<<<1024, 256, 0, stream>>>(Wk + (size_t)l*Hc*Hc, Hc, Hc,  wkT, Hp, Hp, 1);
        prep_weight_kernel<<<1024, 256, 0, stream>>>(Wv + (size_t)l*Hc*Hc, Hc, Hc,  wvT, Hp, Hp, 1);
        prep_weight_kernel<<<1024, 256, 0, stream>>>(Wo + (size_t)l*Hc*Hc, Hc, Hc,  woT, Hp, Hp, 0);
        prep_weight_kernel<<<4096, 256, 0, stream>>>(W1 + (size_t)l*Hc*F4, Hc, F4,  w1T, F4p, Hp, 0);
        prep_weight_kernel<<<4096, 256, 0, stream>>>(W2 + (size_t)l*F4*Hc, F4, Hc,  w2T, Hp, F4p, 0);
    }

    // ---- node assembly ----
    build_nodes_kernel<<<(BSZ * Nc * Hp) / 256, 256, 0, stream>>>(
        ent_vec, entity_num, rels, rel_embed, Af, Bh);

    for (int l = 0; l < 2; ++l) {
        unsigned short* wqT = (unsigned short*)(ws + OFF_W + (size_t)l * 6291456);
        unsigned short* wkT = wqT + 262144;
        unsigned short* wvT = wkT + 262144;
        unsigned short* woT = wvT + 262144;
        unsigned short* w1T = woT + 262144;
        unsigned short* w2T = w1T + 1048576;

        // Q, K (head-padded columns, pads zero via weight pads)
        gemm(stream, Bh,0,0,Hp,  wqT,0,0,Hp,  nullptr,nullptr, Ch,0,0,Hp, MTOT,512,512,512, 1,1, MODE_BF16);
        gemm(stream, Bh,0,0,Hp,  wkT,0,0,Hp,  nullptr,nullptr, Dh,0,0,Hp, MTOT,512,512,512, 1,1, MODE_BF16);
        // V stored transposed per (b,head): vT[b][h][d][i]
        gemm(stream, Bh,0,0,Hp,  wvT,0,0,Hp,  nullptr,nullptr, Eh,0,0,0,  MTOT,512,512,512, 1,1, MODE_BF16_VT);

        // scores: 128 batched (b,head): s = q . k^T   (K = dh padded 128)
        gemm(stream, Ch,262144,128,Hp,  Dh,262144,128,Hp,  nullptr,nullptr,
             Ff,1048576,262144,512, 512,512,512,128, 4,128, MODE_F32);

        // mask + softmax -> bf16 probs
        softmax_mask_kernel<<<BSZ * NH * Nc, 256, 0, stream>>>(Ff, adj, Gh);

        // ctx: 128 batched: a @ v  -> ctx bf16 (column offset head*125)
        gemm(stream, Gh,1048576,262144,512,  Eh,262144,65536,512,  nullptr,nullptr,
             Ch,262144,125,Hp, 512,125,128,512, 4,128, MODE_BF16);

        // o = ctx @ Wo + bo  (stale ctx pad cols killed by zero weight pads)
        gemm(stream, Ch,0,0,Hp,  woT,0,0,Hp,  bo + (size_t)l*Hc, nullptr,
             Ff,0,0,Hc, MTOT,Hc,512,512, 1,1, MODE_F32);

        // t = LN1(o) -> Af (fp32) + Bh (bf16 padded)
        layernorm_kernel<<<MTOT, 512, 0, stream>>>(Ff, nullptr,
            ln1_g + (size_t)l*Hc, ln1_b + (size_t)l*Hc, Af, Bh);

        // h1 = PReLU(t @ W1 + b1) -> Gh bf16 (2048-padded, zero pads)
        gemm(stream, Bh,0,0,Hp,  w1T,0,0,Hp,  b1 + (size_t)l*F4, prelu_a + (size_t)l*F4,
             Gh,0,0,F4p, MTOT,F4,F4p,512, 1,1, MODE_BF16_PRELU);

        // f = h1 @ W2 + b2 -> Ff fp32
        gemm(stream, Gh,0,0,F4p,  w2T,0,0,F4p,  b2 + (size_t)l*Hc, nullptr,
             Ff,0,0,Hc, MTOT,Hc,512,F4p, 1,1, MODE_F32);

        // nodes = LN2(f + t) -> Af (fp32) + Bh (bf16 for next layer)
        layernorm_kernel<<<MTOT, 512, 0, stream>>>(Ff, Af,
            ln2_g + (size_t)l*Hc, ln2_b + (size_t)l*Hc, Af, Bh);
    }

    // ---- outputs ----
    node_emb_kernel<<<(BSZ * Nc * Hc + 255) / 256, 256, 0, stream>>>(Af, entity_num, out);
    glob_mask_kernel<<<(BSZ * Nc + 255) / 256, 256, 0, stream>>>(Af, entity_num, out);
}
